// MultiHeadAttention_19284403159733
// MI455X (gfx1250) — matrix operand order repaired and checked by reading, not compiled, not run on hardware
//
#include <hip/hip_runtime.h>

typedef __attribute__((ext_vector_type(16))) _Float16 v16h;
typedef __attribute__((ext_vector_type(8)))  _Float16 v8h;
typedef __attribute__((ext_vector_type(8)))  float    v8f;

#define WMMA_F16(a, b, c) \
  __builtin_amdgcn_wmma_f32_16x16x32_f16(false, (a), false, (b), (short)0, (c), false, false)

constexpr int BATCH = 4, SEQ = 2048, DIM = 768, HEADS = 16, HD = 48, HDP = 64;
constexpr int HDV = 64;                             // Vt rows: 48 data + ones row + pad
constexpr int TOK = BATCH * SEQ;                    // 8192
constexpr float SCALE = 0.14433756729740643f;       // 48^-0.5

static __device__ __forceinline__ v16h make_a(v8h lo, v8h hi) {
  return __builtin_shufflevector(lo, hi, 0, 1, 2, 3, 4, 5, 6, 7,
                                 8, 9, 10, 11, 12, 13, 14, 15);
}

// DPP16 row_ror rotation within a 16-lane row (wave32 halves = D-layout groups).
template <int CTRL>
static __device__ __forceinline__ float dpp_ror(float v) {
  return __int_as_float(__builtin_amdgcn_update_dpp(
      0, __float_as_int(v), CTRL, 0xf, 0xf, true));
}
static __device__ __forceinline__ float rowmax16(float v) {
  v = fmaxf(v, dpp_ror<0x128>(v));   // row_ror:8
  v = fmaxf(v, dpp_ror<0x124>(v));   // row_ror:4
  v = fmaxf(v, dpp_ror<0x122>(v));   // row_ror:2
  v = fmaxf(v, dpp_ror<0x121>(v));   // row_ror:1
  return v;
}

// ---------------------------------------------------------------------------
// Kernel 1: fp32 -> f16 of x and the 4 weights; zero Hd pad cols of Qp/Kp;
// write the ones row (48) and zero rows 49..63 of Vt. All rewritten each call.
// ---------------------------------------------------------------------------
__global__ void convert_pack_kernel(const float* __restrict__ x,
                                    const float* __restrict__ wq,
                                    const float* __restrict__ wk,
                                    const float* __restrict__ wv,
                                    const float* __restrict__ wo,
                                    _Float16* __restrict__ Xh,
                                    _Float16* __restrict__ Wh,
                                    _Float16* __restrict__ Qp,
                                    _Float16* __restrict__ Kp,
                                    _Float16* __restrict__ Vt) {
  size_t i = (size_t)blockIdx.x * blockDim.x + threadIdx.x;
  if (i < (size_t)TOK * DIM) Xh[i] = (_Float16)x[i];
  if (i < (size_t)4 * DIM * DIM) {
    int j = (int)(i / (DIM * DIM));
    int r = (int)(i % (DIM * DIM));
    const float* w = (j == 0) ? wq : (j == 1) ? wk : (j == 2) ? wv : wo;
    Wh[i] = (_Float16)w[r];
  }
  if (i < (size_t)BATCH * HEADS * SEQ * 16) {  // 64*2048*16 pad elements
    int bh  = (int)(i >> 15);
    int rem = (int)(i & 32767);
    int s = rem >> 4;                 // token / key index
    int j = rem & 15;                 // pad lane 0..15
    size_t qkidx = (((size_t)bh << 11) + s) * HDP + (HD + j);
    Qp[qkidx] = (_Float16)0.f;
    Kp[qkidx] = (_Float16)0.f;
    // Vt row 48 = ones (row-sum column for softmax denominator), 49..63 = 0
    Vt[(((size_t)bh * HDV) + 48 + j) * SEQ + s] = (j == 0) ? (_Float16)1.f
                                                           : (_Float16)0.f;
  }
}

// ---------------------------------------------------------------------------
// Kernel 2: Y = X @ W^T + b. One wave computes a 32x64 output block:
// 2 A-fragments x 4 B-fragments -> 8 chained v_wmma_f32_16x16x32_f16 per
// 32-wide K step (24 steps over K=768).
//   MODE 0: f32 row-major to out    MODE 1/2: f16 scatter to Qp/Kp [bh][s][64]
//   MODE 3: f16 scatter to Vt [bh][d][2048]
// ---------------------------------------------------------------------------
template <int MODE>
__global__ void gemm16_kernel(const _Float16* __restrict__ X,
                              const _Float16* __restrict__ W,
                              const float* __restrict__ bias,
                              void* __restrict__ out) {
  const int wave = threadIdx.x >> 5, lane = threadIdx.x & 31;
  const int lh = lane >> 4, lm = lane & 15;
  const int tile = blockIdx.x * (blockDim.x >> 5) + wave;  // 256*12 tiles exact
  const int mt = tile / (DIM / 64), nt = tile % (DIM / 64);
  const int m0 = mt * 32, n0 = nt * 64;

  const _Float16* xrow0 = X + (size_t)(m0 + lm) * DIM;
  const _Float16* xrow1 = X + (size_t)(m0 + 16 + lm) * DIM;
  const _Float16* wrow  = W + (size_t)(n0 + lm) * DIM;

  v8f acc[2][4] = {};
  for (int kb = 0; kb < DIM; kb += 32) {
    __builtin_prefetch(xrow0 + kb + 128, 0, 3);   // global_prefetch_b8
    v16h a0 = make_a(*(const v8h*)(xrow0 + kb + lh * 8),
                     *(const v8h*)(xrow0 + kb + 16 + lh * 8));
    v16h a1 = make_a(*(const v8h*)(xrow1 + kb + lh * 8),
                     *(const v8h*)(xrow1 + kb + 16 + lh * 8));
#pragma unroll
    for (int j = 0; j < 4; ++j) {
      v16h b = make_a(*(const v8h*)(wrow + (size_t)j * 16 * DIM + kb + lh * 8),
                      *(const v8h*)(wrow + (size_t)j * 16 * DIM + kb + 16 + lh * 8));
      acc[0][j] = WMMA_F16(a0, b, acc[0][j]);
      acc[1][j] = WMMA_F16(a1, b, acc[1][j]);
    }
  }
#pragma unroll
  for (int j = 0; j < 4; ++j) {
    const int n = n0 + j * 16 + lm;
    const float bn = bias[n];
    const int h = n / HD, d = n - h * HD;
#pragma unroll
    for (int i = 0; i < 2; ++i) {
#pragma unroll
      for (int r = 0; r < 8; ++r) {
        float val = acc[i][j][r] + bn;  // D element (M = m0+i*16+8*lh+r, N = n)
        int mm = m0 + i * 16 + 8 * lh + r;
        if (MODE == 0) {
          ((float*)out)[(size_t)mm * DIM + n] = val;
        } else {
          int bidx = mm >> 11, s = mm & (SEQ - 1);
          int bh = bidx * HEADS + h;
          if (MODE == 3) {  // Vt[bh][d][s]
            ((_Float16*)out)[(((size_t)bh * HDV + d) << 11) + s] = (_Float16)val;
          } else {          // Qp/Kp[bh][s][64]
            ((_Float16*)out)[((((size_t)bh << 11) + s) << 6) + d] = (_Float16)val;
          }
        }
      }
    }
  }
}

// ---------------------------------------------------------------------------
// Kernel 3: fused flash attention. One wave per (bh, 16-query tile), 32 keys
// per step: 4 score WMMAs (K-dim padded to 64), online softmax (row max via
// DPP row_ror butterfly; row sum comes FREE from a 4th P@V accumulator against
// the ones-row of Vt), P staged through LDS (D->A relayout), 4 P@V WMMAs.
// ---------------------------------------------------------------------------
__global__ void attn_kernel(const _Float16* __restrict__ Qp,
                            const _Float16* __restrict__ Kp,
                            const _Float16* __restrict__ Vt,
                            _Float16* __restrict__ Ao) {
  __shared__ alignas(32) _Float16 pbuf[8][16][32];  // 1KB per wave
  const int wave = threadIdx.x >> 5, lane = threadIdx.x & 31;
  const int lh = lane >> 4, lm = lane & 15;
  const int tile = blockIdx.x * 8 + wave;  // 64*128 tiles exact
  const int bh = tile >> 7, qt = tile & 127;

  const _Float16* qrow = Qp + (((size_t)bh << 11) + qt * 16 + lm) * HDP;
  v16h a_lo = *(const v16h*)(qrow + lh * 16);
  v16h a_hi = *(const v16h*)(qrow + 32 + lh * 16);

  v8f acc0 = {}, acc1 = {}, acc2 = {}, acc3 = {};  // acc3: row-sum column
  float mrow[8];
#pragma unroll
  for (int r = 0; r < 8; ++r) mrow[r] = -1e30f;

  const _Float16* vt = Vt + ((size_t)bh * HDV) * SEQ;
  for (int kb = 0; kb < SEQ; kb += 32) {
    const _Float16* k0 = Kp + (((size_t)bh << 11) + kb + lm) * HDP;
    const _Float16* k1 = k0 + 16 * HDP;
    v8f z = {};
    v8f s0 = WMMA_F16(a_hi, *(const v16h*)(k0 + 32 + lh * 16),
                      WMMA_F16(a_lo, *(const v16h*)(k0 + lh * 16), z));
    v8f s1 = WMMA_F16(a_hi, *(const v16h*)(k1 + 32 + lh * 16),
                      WMMA_F16(a_lo, *(const v16h*)(k1 + lh * 16), z));
    float alpha[8];
#pragma unroll
    for (int r = 0; r < 8; ++r) {
      float x0 = s0[r] * SCALE, x1 = s1[r] * SCALE;
      float mx = rowmax16(fmaxf(x0, x1));          // DPP butterfly, 16 lanes
      float mnew = fmaxf(mrow[r], mx);
      alpha[r] = __expf(mrow[r] - mnew);
      mrow[r] = mnew;
      float p0 = __expf(x0 - mnew), p1 = __expf(x1 - mnew);
      pbuf[wave][8 * lh + r][lm]      = (_Float16)p0;  // D layout -> LDS
      pbuf[wave][8 * lh + r][16 + lm] = (_Float16)p1;
      acc0[r] *= alpha[r];
      acc1[r] *= alpha[r];
      acc2[r] *= alpha[r];
      acc3[r] *= alpha[r];
    }
    asm volatile("s_wait_dscnt 0" ::: "memory");  // same-wave DS in order
    v16h pa = *(const v16h*)(&pbuf[wave][lm][lh * 16]);
    const _Float16* vbase = vt + kb + lh * 16;
    acc0 = WMMA_F16(pa, *(const v16h*)(vbase + ((size_t)lm << 11)),        acc0);
    acc1 = WMMA_F16(pa, *(const v16h*)(vbase + ((size_t)(16 + lm) << 11)), acc1);
    acc2 = WMMA_F16(pa, *(const v16h*)(vbase + ((size_t)(32 + lm) << 11)), acc2);
    acc3 = WMMA_F16(pa, *(const v16h*)(vbase + ((size_t)(48 + lm) << 11)), acc3);
  }

  const int b = bh >> 4, h = bh & 15;
#pragma unroll
  for (int r = 0; r < 8; ++r) {
    // softmax denominator lives in column N=0 of acc3 (ones row of Vt):
    float lsum = __shfl(acc3[r], lh * 16, 32);
    float inv = 1.0f / lsum;
    int s = qt * 16 + 8 * lh + r;
    size_t base = ((size_t)(b * SEQ + s)) * DIM + h * HD;
    Ao[base + lm]      = (_Float16)(acc0[r] * inv);
    Ao[base + 16 + lm] = (_Float16)(acc1[r] * inv);
    Ao[base + 32 + lm] = (_Float16)(acc2[r] * inv);
  }
}

// ---------------------------------------------------------------------------
extern "C" void kernel_launch(void* const* d_in, const int* in_sizes, int n_in,
                              void* d_out, int out_size, void* d_ws, size_t ws_size,
                              hipStream_t stream) {
  (void)in_sizes; (void)n_in; (void)out_size; (void)ws_size;
  const float* x  = (const float*)d_in[0];
  const float* Wq = (const float*)d_in[1];
  const float* bq = (const float*)d_in[2];
  const float* Wk = (const float*)d_in[3];
  const float* bk = (const float*)d_in[4];
  const float* Wv = (const float*)d_in[5];
  const float* bv = (const float*)d_in[6];
  const float* Wo = (const float*)d_in[7];
  const float* bo = (const float*)d_in[8];

  char* ws = (char*)d_ws;
  size_t off = 0;
  auto take = [&](size_t bytes) -> char* {
    char* p = ws + off;
    off += (bytes + 255) & ~(size_t)255;
    return p;
  };
  _Float16* Xh = (_Float16*)take((size_t)TOK * DIM * 2);                 // 12.6 MB
  _Float16* Wh = (_Float16*)take((size_t)4 * DIM * DIM * 2);             //  4.7 MB
  _Float16* Qp = (_Float16*)take((size_t)BATCH * HEADS * SEQ * HDP * 2); // 16.8 MB
  _Float16* Kp = (_Float16*)take((size_t)BATCH * HEADS * SEQ * HDP * 2); // 16.8 MB
  _Float16* Vt = (_Float16*)take((size_t)BATCH * HEADS * HDV * SEQ * 2); // 16.8 MB
  _Float16* Ao = (_Float16*)take((size_t)TOK * DIM * 2);                 // 12.6 MB

  {
    size_t N = (size_t)TOK * DIM;  // covers all guarded loops
    int blocks = (int)((N + 255) / 256);
    convert_pack_kernel<<<blocks, 256, 0, stream>>>(x, Wq, Wk, Wv, Wo,
                                                    Xh, Wh, Qp, Kp, Vt);
  }
  const int gemm_blocks = (TOK / 32) * (DIM / 64) / 8;  // 384
  gemm16_kernel<1><<<gemm_blocks, 256, 0, stream>>>(Xh, Wh + 0 * (size_t)DIM * DIM, bq, Qp);
  gemm16_kernel<2><<<gemm_blocks, 256, 0, stream>>>(Xh, Wh + 1 * (size_t)DIM * DIM, bk, Kp);
  gemm16_kernel<3><<<gemm_blocks, 256, 0, stream>>>(Xh, Wh + 2 * (size_t)DIM * DIM, bv, Vt);

  attn_kernel<<<(BATCH * HEADS * (SEQ / 16)) / 8, 256, 0, stream>>>(Qp, Kp, Vt, Ao);

  gemm16_kernel<0><<<gemm_blocks, 256, 0, stream>>>(Ao, Wh + 3 * (size_t)DIM * DIM, bo, d_out);
}
